// HeteGraphAttentionLayer_47888885350604
// MI455X (gfx1250) — matrix operand order repaired and checked by reading, not compiled, not run on hardware
//
#include <hip/hip_runtime.h>
#include <hip/hip_bf16.h>

typedef __attribute__((ext_vector_type(16))) __bf16 v16bf;
typedef __attribute__((ext_vector_type(8)))  __bf16 v8bf;
typedef __attribute__((ext_vector_type(8)))  float  v8f;
typedef __attribute__((ext_vector_type(4)))  float  v4f;

#define B_ 8
#define N_ 1024
#define D_ 256

__device__ __forceinline__ __bf16 f2bf(float f) {
    unsigned u = __builtin_bit_cast(unsigned, f);
    unsigned r = u + 0x7FFFu + ((u >> 16) & 1u);        // round-to-nearest-even
    unsigned short h = (unsigned short)(r >> 16);
    return __builtin_bit_cast(__bf16, h);
}
__device__ __forceinline__ float bf2f(__bf16 b) {
    unsigned short h = __builtin_bit_cast(unsigned short, b);
    unsigned u = ((unsigned)h) << 16;
    return __builtin_bit_cast(float, u);
}

__device__ __forceinline__ v16bf ldA(const __bf16* p) {
    // A fragment halves: K = half*16+{0..15} (contiguous)
    v8bf lo = *(const v8bf*)(p);
    v8bf hi = *(const v8bf*)(p + 8);
    return __builtin_shufflevector(lo, hi,
        0, 1, 2, 3, 4, 5, 6, 7, 8, 9, 10, 11, 12, 13, 14, 15);
}

// -------- pre-pass: f32 -> bf16, 8 elements/thread, fully coalesced --------
__global__ __launch_bounds__(256)
void k_f32_to_bf16(const float* __restrict__ src, __bf16* __restrict__ dst, int n8) {
    const int idx = blockIdx.x * blockDim.x + threadIdx.x;
    if (idx >= n8) return;
    v4f x0 = *(const v4f*)(src + (size_t)idx * 8);
    v4f x1 = *(const v4f*)(src + (size_t)idx * 8 + 4);
    v8bf o;
#pragma unroll
    for (int e = 0; e < 4; ++e) { o[e] = f2bf(x0[e]); o[4 + e] = f2bf(x1[e]); }
    *(v8bf*)(dst + (size_t)idx * 8) = o;
}

// -------- GEMM1: h = X @ W^T (bf16 in, f32 acc), write hT[b][o][n] bf16 --------
// grid = B * N/16 blocks, 128 threads (4 waves); wave w covers feature cols w*64..+63
// Software-pipelined: fragments for kt+1 are loaded before the WMMAs of kt.
__global__ __launch_bounds__(128)
void k_gemm1(const __bf16* __restrict__ Xb, const __bf16* __restrict__ Wb,
             __bf16* __restrict__ hT) {
    const int wave = threadIdx.x >> 5;
    const int lane = threadIdx.x & 31;
    const int m    = lane & 15;
    const int half = lane >> 4;
    const int b    = blockIdx.x / (N_ / 16);
    const int n0   = (blockIdx.x % (N_ / 16)) * 16;
    const __bf16* Ap = Xb + ((size_t)b * N_ + (n0 + m)) * D_ + half * 16;
    const int obase = wave * 64;
    const __bf16* Bp[4];
#pragma unroll
    for (int t = 0; t < 4; ++t)
        Bp[t] = Wb + (size_t)(obase + t * 16 + m) * D_ + half * 16;

    v8f acc[4] = {};
    v16bf a_cur = ldA(Ap);
    v16bf b_cur[4];
#pragma unroll
    for (int t = 0; t < 4; ++t) b_cur[t] = *(const v16bf*)(Bp[t]);

#pragma unroll
    for (int kt = 0; kt < D_ / 32; ++kt) {
        v16bf a_nxt;
        v16bf b_nxt[4];
        if (kt < D_ / 32 - 1) {                      // static under full unroll
            const int kn = (kt + 1) * 32;
            a_nxt = ldA(Ap + kn);
#pragma unroll
            for (int t = 0; t < 4; ++t) b_nxt[t] = *(const v16bf*)(Bp[t] + kn);
        }
#pragma unroll
        for (int t = 0; t < 4; ++t)
            acc[t] = __builtin_amdgcn_wmma_f32_16x16x32_bf16(
                false, a_cur, false, b_cur[t], (short)0, acc[t], false, false);
        if (kt < D_ / 32 - 1) {
            a_cur = a_nxt;
#pragma unroll
            for (int t = 0; t < 4; ++t) b_cur[t] = b_nxt[t];
        }
    }
    // D layout: lane(col=m, hh=half), VGPR r -> row r+8*hh. Transposed store contiguous.
#pragma unroll
    for (int t = 0; t < 4; ++t) {
        const int o = obase + t * 16 + m;
        v8bf pk;
#pragma unroll
        for (int r = 0; r < 8; ++r) pk[r] = f2bf(acc[t][r]);
        *(v8bf*)(hT + ((size_t)b * D_ + o) * N_ + n0 + 8 * half) = pk;
    }
}

// -------- scores: s1[k,b,n] = sum_o hT[b][o][n]*a1[k][o]  (and s2) --------
__global__ __launch_bounds__(256)
void k_scores(const __bf16* __restrict__ hT, const float* __restrict__ a1,
              const float* __restrict__ a2, float* __restrict__ s1,
              float* __restrict__ s2) {
    const int idx = blockIdx.x * blockDim.x + threadIdx.x;  // b*N + n
    const int b = idx >> 10, n = idx & (N_ - 1);
    const __bf16* base = hT + (size_t)b * D_ * N_ + n;
    float acc1[4] = {0.f, 0.f, 0.f, 0.f}, acc2[4] = {0.f, 0.f, 0.f, 0.f};
    for (int o = 0; o < D_; ++o) {
        float v = bf2f(base[(size_t)o * N_]);   // coalesced across n
#pragma unroll
        for (int k = 0; k < 4; ++k) {
            acc1[k] += v * a1[k * D_ + o];      // uniform -> scalar loads
            acc2[k] += v * a2[k * D_ + o];
        }
    }
#pragma unroll
    for (int k = 0; k < 4; ++k) {
        s1[((k * B_ + b) << 10) + n] = acc1[k];
        s2[((k * B_ + b) << 10) + n] = acc2[k];
    }
}

// -------- logits + adj mask + row softmax -> P (bf16). One wave32 per row. --------
__global__ __launch_bounds__(256)
void k_attn_softmax(const float* __restrict__ s1, const float* __restrict__ s2,
                    const int* __restrict__ adj, const int* __restrict__ seqp,
                    const int* __restrict__ segp, __bf16* __restrict__ P) {
    const int wave = threadIdx.x >> 5;
    const int lane = threadIdx.x & 31;
    const int row  = blockIdx.x * 8 + wave;
    const int b = row >> 10, i = row & (N_ - 1);
    const int seq = *seqp, hi = seq + *segp;

    float s1k[4];
#pragma unroll
    for (int k = 0; k < 4; ++k) s1k[k] = s1[((k * B_ + b) << 10) + i];
    const int* adjr = adj + ((size_t)b * N_ + i) * N_;
    const bool iA = i < seq, iS = (i >= seq) & (i < hi), iG = i >= hi;

    float vals[32];
    float mx = -3.4e38f;
#pragma unroll
    for (int jj = 0; jj < 32; ++jj) {
        const int j = jj * 32 + lane;
        const bool jA = j < seq, jS = (j >= seq) & (j < hi), jG = j >= hi;
        int k = -1;
        if (i == j)                         k = 0;   // m0: identity
        else if (iS & jS)                   k = 1;   // m1: seg block, off-diag
        else if ((iA & jS) | (iS & jA))     k = 2;   // m2: seq <-> seg
        else if ((iG & jA) | (iA & jG))     k = 3;   // m3: gaz <-> seq
        float v = 0.f;
        if (k >= 0) {
            float e = s1k[k] + s2[((k * B_ + b) << 10) + j];
            v = e > 0.f ? e : 0.2f * e;              // LeakyReLU(0.2)
        }
        if (adjr[j] == 0) v = -1e9f;
        vals[jj] = v;
        mx = fmaxf(mx, v);
    }
#pragma unroll
    for (int off = 16; off > 0; off >>= 1) mx = fmaxf(mx, __shfl_xor(mx, off, 32));
    float sum = 0.f;
#pragma unroll
    for (int jj = 0; jj < 32; ++jj) { vals[jj] = __expf(vals[jj] - mx); sum += vals[jj]; }
#pragma unroll
    for (int off = 16; off > 0; off >>= 1) sum += __shfl_xor(sum, off, 32);
    const float inv = 1.f / sum;
    __bf16* Pr = P + ((size_t)b * N_ + i) * N_;
#pragma unroll
    for (int jj = 0; jj < 32; ++jj) Pr[jj * 32 + lane] = f2bf(vals[jj] * inv);
}

// -------- GEMM2: out = elu(P @ h); P bf16 [b][i][j], h via hT bf16 [b][o][j] --------
// grid = B * N/32, 256 threads (8 waves). Wave: row-tile (w>>2), col-tiles (w&3)*4..+3
// Software-pipelined K loop: branch-free prologue / steady state / epilogue.
__global__ __launch_bounds__(256)
void k_gemm2(const __bf16* __restrict__ P, const __bf16* __restrict__ hT,
             float* __restrict__ out) {
    const int wave = threadIdx.x >> 5;
    const int lane = threadIdx.x & 31;
    const int m    = lane & 15;
    const int half = lane >> 4;
    const int b    = blockIdx.x / (N_ / 32);
    const int i0   = (blockIdx.x % (N_ / 32)) * 32 + (wave >> 2) * 16;
    const int ct0  = (wave & 3) * 4;
    const __bf16* Ap = P + ((size_t)b * N_ + (i0 + m)) * N_ + half * 16;
    const __bf16* Bp[4];
#pragma unroll
    for (int t = 0; t < 4; ++t)
        Bp[t] = hT + ((size_t)b * D_ + (ct0 + t) * 16 + m) * N_ + half * 16;

    v8f acc[4] = {};
    v16bf a_cur = ldA(Ap);
    v16bf b_cur[4];
#pragma unroll
    for (int t = 0; t < 4; ++t) b_cur[t] = *(const v16bf*)(Bp[t]);

#pragma unroll 4
    for (int kt = 0; kt < N_ / 32 - 1; ++kt) {
        const int kn = (kt + 1) * 32;
        v16bf a_nxt = ldA(Ap + kn);
        v16bf b_nxt[4];
#pragma unroll
        for (int t = 0; t < 4; ++t) b_nxt[t] = *(const v16bf*)(Bp[t] + kn);
#pragma unroll
        for (int t = 0; t < 4; ++t)
            acc[t] = __builtin_amdgcn_wmma_f32_16x16x32_bf16(
                false, a_cur, false, b_cur[t], (short)0, acc[t], false, false);
        a_cur = a_nxt;
#pragma unroll
        for (int t = 0; t < 4; ++t) b_cur[t] = b_nxt[t];
    }
#pragma unroll
    for (int t = 0; t < 4; ++t)      // epilogue: last K step
        acc[t] = __builtin_amdgcn_wmma_f32_16x16x32_bf16(
            false, a_cur, false, b_cur[t], (short)0, acc[t], false, false);

#pragma unroll
    for (int t = 0; t < 4; ++t) {
        const int o = (ct0 + t) * 16 + m;
#pragma unroll
        for (int r = 0; r < 8; ++r) {
            const int i = i0 + r + 8 * half;
            float v = acc[t][r];
            v = v > 0.f ? v : (__expf(v) - 1.f);    // ELU
            out[((size_t)b * N_ + i) * D_ + o] = v;
        }
    }
}

extern "C" void kernel_launch(void* const* d_in, const int* in_sizes, int n_in,
                              void* d_out, int out_size, void* d_ws, size_t ws_size,
                              hipStream_t stream) {
    const float* X   = (const float*)d_in[0];
    const int*   adj = (const int*)d_in[1];
    const float* W   = (const float*)d_in[2];
    const float* a1  = (const float*)d_in[3];
    const float* a2  = (const float*)d_in[4];
    const int* seqp  = (const int*)d_in[5];
    const int* segp  = (const int*)d_in[6];
    float* out = (float*)d_out;

    char* w = (char*)d_ws;
    __bf16* Xb = (__bf16*)(w);                     // 8*1024*256*2 = 4 MiB
    __bf16* Wb = (__bf16*)(w + 0x400000);          // 256*256*2    = 128 KiB
    __bf16* hT = (__bf16*)(w + 0x420000);          // 4 MiB
    float*  s1 = (float*)(w + 0x820000);           // 128 KiB
    float*  s2 = (float*)(w + 0x840000);           // 128 KiB
    __bf16* P  = (__bf16*)(w + 0x860000);          // 16 MiB

    const int nX8 = (B_ * N_ * D_) / 8;            // 262144
    const int nW8 = (D_ * D_) / 8;                 // 8192
    hipLaunchKernelGGL(k_f32_to_bf16, dim3(nX8 / 256), dim3(256), 0, stream, X, Xb, nX8);
    hipLaunchKernelGGL(k_f32_to_bf16, dim3(nW8 / 256), dim3(256), 0, stream, W, Wb, nW8);
    hipLaunchKernelGGL(k_gemm1, dim3(B_ * (N_ / 16)), dim3(128), 0, stream, Xb, Wb, hT);
    hipLaunchKernelGGL(k_scores, dim3((B_ * N_) / 256), dim3(256), 0, stream,
                       hT, a1, a2, s1, s2);
    hipLaunchKernelGGL(k_attn_softmax, dim3((B_ * N_) / 8), dim3(256), 0, stream,
                       s1, s2, adj, seqp, segp, P);
    hipLaunchKernelGGL(k_gemm2, dim3(B_ * (N_ / 32)), dim3(256), 0, stream, P, hT, out);
}
